// MultiHeadedAttentionTimeRestricted_75436805587618
// MI455X (gfx1250) — matrix operand order repaired and checked by reading, not compiled, not run on hardware
//
#include <hip/hip_runtime.h>
#include <hip/hip_bf16.h>

#define B_ 4
#define T_ 1024
#define W_ 33
#define F_ 512
#define H_ 8
#define DK_ 64

typedef __attribute__((ext_vector_type(16))) __bf16 v16bf;
typedef __attribute__((ext_vector_type(8)))  float  v8f;
typedef __attribute__((ext_vector_type(2)))  float  v2f;
typedef __attribute__((ext_vector_type(2)))  __bf16 v2bf;

// native bf16 converts (gfx1250; fptrunc is RNE). Vector form -> v_cvt_pk_bf16_f32.
__device__ __forceinline__ unsigned short f2bf(float x) {
  union { __bf16 h; unsigned short u; } t;
  t.h = (__bf16)x;
  return t.u;
}
__device__ __forceinline__ unsigned pack2bf(float x, float y) {
  v2f a = {x, y};
  union { v2bf h; unsigned u; } t;
  t.h = __builtin_convertvector(a, v2bf);
  return t.u;
}
__device__ __forceinline__ float bf2f(unsigned short u) {
  union { __bf16 h; unsigned short u_; } t;
  t.u_ = u;
  return (float)t.h;
}

// ---- CDNA5 async global->LDS copy (ASYNCcnt path, §15.18 / 08_async_tensor) ----
__device__ __forceinline__ void async_copy_b128(unsigned lds_off, unsigned long long gaddr) {
  asm volatile("global_load_async_to_lds_b128 %0, %1, off"
               :: "v"(lds_off), "v"(gaddr) : "memory");
}
__device__ __forceinline__ void wait_async0() {
  asm volatile("s_wait_asynccnt 0x0" ::: "memory");
}
__device__ __forceinline__ void wait_ds0() {
  asm volatile("s_wait_dscnt 0x0" ::: "memory");
}
__device__ __forceinline__ unsigned lds_addr32(const void* p) {
  return (unsigned)(unsigned long long)p;   // low 32 bits of generic addr == LDS offset
}

union FragB16 { unsigned u[8]; v16bf v; };

// A fragment (16x32 bf16, M x K) from LDS row-major [rows][SA] (rows pre-zero-padded)
template <int SA>
__device__ __forceinline__ v16bf load_frag_a(const unsigned short* buf, int m0, int k0, int lane) {
  const int half = lane >> 4, r = lane & 15;
  const int row = m0 + r;
  FragB16 f;
#pragma unroll
  for (int j = 0; j < 8; ++j) {
    const int kk = k0 + 16 * half + 2 * j;
    f.u[j] = *(const unsigned*)(buf + row * SA + kk);
  }
  return f.v;
}

// B fragment (32x16 bf16, K x N) from LDS weight chunk stored transposed [N][SB]
template <int SB>
__device__ __forceinline__ v16bf load_frag_b(const unsigned short* buf, int n0, int k0, int lane) {
  const int half = lane >> 4, r = lane & 15;
  const unsigned short* p = buf + (n0 + r) * SB + k0 + 16 * half;
  FragB16 f;
#pragma unroll
  for (int j = 0; j < 8; ++j) f.u[j] = *(const unsigned*)(p + 2 * j);
  return f.v;
}

// ---------- weight fp32 -> bf16, pre-transposed chunk layout [kc][n][kk] ----------
// Output element ((k>>5)*512 + n)*32 + (k&31) = W[k][n]; each k-chunk row (n) is a
// contiguous 64B run -> consumers stage it with pure async b128 copies.
__global__ void cvt_transpose(const float* __restrict__ s, unsigned short* __restrict__ d) {
  for (int i = blockIdx.x * blockDim.x + threadIdx.x; i < F_ * F_; i += gridDim.x * blockDim.x) {
    const int k = i >> 9, n = i & 511;
    d[(((k >> 5) * F_) + n) * 32 + (k & 31)] = f2bf(s[i]);
  }
}

// ---------------- generic GEMM: C[M,512] = A[M,512] @ W + bias ----------------
// block tile 64x64, 8 waves (4 M-tiles x 2 N-tile-pairs), grid = (M/64, 512/64)
__launch_bounds__(256)
__global__ void gemm_bias(const float* __restrict__ A, const unsigned short* __restrict__ WT,
                          const float* __restrict__ bias, float* __restrict__ C) {
  __shared__ __align__(16) unsigned short Ash[64 * 36];
  __shared__ __align__(16) unsigned short Bsh[64 * 40];
  const int tid = threadIdx.x, lane = tid & 31, wid = tid >> 5;
  const int mb = blockIdx.x * 64, nb = blockIdx.y * 64;
  const int m0 = (wid & 3) * 16;
  const int n0 = (wid >> 2) * 32;
  v8f acc0 = {0, 0, 0, 0, 0, 0, 0, 0};
  v8f acc1 = {0, 0, 0, 0, 0, 0, 0, 0};

  for (int kc = 0; kc < F_ / 32; ++kc) {
    const int k0 = kc * 32;
    __syncthreads();
    {   // async-stage weight chunk rows [nb, nb+64): one b128 per thread
      const int n = tid >> 2, q = tid & 3;
      const unsigned long long g =
          (unsigned long long)(const void*)WT +
          ((unsigned long long)(kc * F_ + nb + n)) * 64ull + (unsigned long long)q * 16ull;
      async_copy_b128(lds_addr32(&Bsh[n * 40 + q * 8]), g);
    }
#pragma unroll
    for (int i = tid; i < 64 * 8; i += 256) {   // stage A 64x32 fp32 -> bf16 via b128
      const int r = i >> 3, c4 = i & 7;
      const float4 x = *(const float4*)&A[(long)(mb + r) * F_ + k0 + c4 * 4];
      unsigned* p = (unsigned*)&Ash[r * 36 + c4 * 4];
      p[0] = pack2bf(x.x, x.y);
      p[1] = pack2bf(x.z, x.w);
    }
    wait_async0();
    __syncthreads();
    const v16bf a  = load_frag_a<36>(Ash, m0, 0, lane);
    const v16bf b0 = load_frag_b<40>(Bsh, n0, 0, lane);
    const v16bf b1 = load_frag_b<40>(Bsh, n0 + 16, 0, lane);
    acc0 = __builtin_amdgcn_wmma_f32_16x16x32_bf16(false, a, false, b0, (short)0, acc0, false, false);
    acc1 = __builtin_amdgcn_wmma_f32_16x16x32_bf16(false, a, false, b1, (short)0, acc1, false, false);
  }

  const int half = lane >> 4, r = lane & 15;
  const int col0 = nb + n0 + r, col1 = col0 + 16;
  const float bv0 = bias[col0], bv1 = bias[col1];
#pragma unroll
  for (int v = 0; v < 8; ++v) {
    const int row = mb + m0 + 8 * half + v;
    C[(long)row * F_ + col0] = acc0[v] + bv0;
    C[(long)row * F_ + col1] = acc1[v] + bv1;
  }
}

// ---------------- fused per-(b,t) kernel ----------------
// One workgroup (8 waves) per (b,t):
//   Kp = key[b,t] @ Wk + bk  (33x512 WMMA, M zero-padded to 48), Vp likewise,
//   then per-head softmax(QK^T/8)@V (wave == head) -> x row in ws.
// Weight chunks are staged into PER-WAVE LDS slices with async b128 copies:
// no per-chunk block barriers, each wave gates only on its own ASYNCcnt/DScnt.
__launch_bounds__(256)
__global__ void fused_attn(const float* __restrict__ key, const float* __restrict__ value,
                           const int* __restrict__ mask,
                           const unsigned short* __restrict__ WkT, const float* __restrict__ bk,
                           const unsigned short* __restrict__ WvT, const float* __restrict__ bv,
                           const float* __restrict__ qp, float* __restrict__ xout) {
  __shared__ __align__(16) unsigned short Asm[48 * 520];      // key/value rows, rows 33..47 == 0
  __shared__ __align__(16) unsigned short Bs[8 * 64 * 40];    // per-wave weight slices
  __shared__ __align__(16) unsigned short Kp[33 * 520];       // projected K (bf16)
  __shared__ __align__(16) unsigned short Vp[33 * 520];       // projected V (bf16)
  __shared__ float qs[F_];
  __shared__ float attn_s[H_][W_];

  const int tid = threadIdx.x, lane = tid & 31, wid = tid >> 5;
  const int bt = blockIdx.x;                         // b*T + t
  const long rowbase = (long)bt * (W_ * F_);
  unsigned short* myB = &Bs[wid * (64 * 40)];        // this wave's slice (64 cols x 32 k)

  // zero the M-padding rows once; both passes only overwrite rows 0..32
  for (int i = tid; i < 15 * (F_ / 4); i += 256) {
    const int row = 33 + (i >> 7), c4 = i & 127;
    unsigned* p = (unsigned*)&Asm[row * 520 + c4 * 4];
    p[0] = 0u;
    p[1] = 0u;
  }

  for (int pass = 0; pass < 2; ++pass) {
    const float* src = pass ? value : key;
    const unsigned short* WT = pass ? WvT : WkT;
    const float* bias = pass ? bv : bk;
    unsigned short* Out = pass ? Vp : Kp;

    __syncthreads();                                  // protect Asm reuse
    const float4* src4 = (const float4*)(src + rowbase);
    for (int i = tid; i < (W_ * F_) / 4; i += 256) {  // stage 33x512 rows, b128 loads
      const float4 x = src4[i];
      const int row = i >> 7, c4 = i & 127;
      unsigned* p = (unsigned*)&Asm[row * 520 + c4 * 4];
      p[0] = pack2bf(x.x, x.y);
      p[1] = pack2bf(x.z, x.w);
    }
    __syncthreads();                                  // Asm visible to all waves

    v8f acc[3][4];
#pragma unroll
    for (int mt = 0; mt < 3; ++mt)
#pragma unroll
      for (int nt = 0; nt < 4; ++nt) acc[mt][nt] = (v8f){0, 0, 0, 0, 0, 0, 0, 0};

    const unsigned long long wtbase = (unsigned long long)(const void*)WT;
    for (int kc = 0; kc < 16; ++kc) {
      const int k0 = kc * 32;
      wait_ds0();                                     // own slice reads complete before overwrite
#pragma unroll
      for (int j = 0; j < 8; ++j) {                   // async-stage own 4KB slice: 8 b128/lane
        const int p = j * 32 + lane;                  // 0..255 16B-pieces
        const int n = p >> 2, q = p & 3;              // local col n, quarter q
        async_copy_b128(lds_addr32(&myB[n * 40 + q * 8]),
                        wtbase + (unsigned long long)((kc * F_ + wid * 64 + n) * 64 + q * 16));
      }
      const v16bf a0 = load_frag_a<520>(Asm, 0,  k0, lane);   // overlaps with copy
      const v16bf a1 = load_frag_a<520>(Asm, 16, k0, lane);
      const v16bf a2 = load_frag_a<520>(Asm, 32, k0, lane);
      wait_async0();
#pragma unroll
      for (int nt = 0; nt < 4; ++nt) {                // wave owns N columns [wid*64, wid*64+64)
        const v16bf bfr = load_frag_b<40>(myB, nt * 16, 0, lane);
        acc[0][nt] = __builtin_amdgcn_wmma_f32_16x16x32_bf16(false, a0, false, bfr, (short)0, acc[0][nt], false, false);
        acc[1][nt] = __builtin_amdgcn_wmma_f32_16x16x32_bf16(false, a1, false, bfr, (short)0, acc[1][nt], false, false);
        acc[2][nt] = __builtin_amdgcn_wmma_f32_16x16x32_bf16(false, a2, false, bfr, (short)0, acc[2][nt], false, false);
      }
    }

    const int half = lane >> 4, r = lane & 15;
#pragma unroll
    for (int nt = 0; nt < 4; ++nt) {                  // C layout: lane->N, VGPR->M
      const int col = wid * 64 + nt * 16 + r;
      const float bc = bias[col];
#pragma unroll
      for (int mt = 0; mt < 3; ++mt)
#pragma unroll
        for (int v = 0; v < 8; ++v) {
          const int row = mt * 16 + 8 * half + v;
          if (row < W_) Out[row * 520 + col] = f2bf(acc[mt][nt][v] + bc);
        }
    }
  }

  for (int i = tid; i < F_; i += 256) qs[i] = qp[(long)bt * F_ + i];
  __syncthreads();                                    // Kp/Vp/qs visible to all waves

  // ---- attention: wave == head; lane handles slot w=lane (+ lane0 handles w=32) ----
  const int h = wid;
  const float NEG = -3.402823466e38f;
  float s0, s1 = NEG;
  {
    float a = 0.f;
#pragma unroll 8
    for (int d = 0; d < DK_; ++d) a += qs[h * DK_ + d] * bf2f(Kp[lane * 520 + h * DK_ + d]);
    s0 = (mask[(long)bt * W_ + lane] == 0) ? NEG : a * 0.125f;   // 1/sqrt(64)
  }
  if (lane == 0) {
    float a = 0.f;
#pragma unroll 8
    for (int d = 0; d < DK_; ++d) a += qs[h * DK_ + d] * bf2f(Kp[32 * 520 + h * DK_ + d]);
    s1 = (mask[(long)bt * W_ + 32] == 0) ? NEG : a * 0.125f;
  }
  float m = fmaxf(s0, s1);
#pragma unroll
  for (int off = 16; off > 0; off >>= 1) m = fmaxf(m, __shfl_xor(m, off, 32));
  float p0 = __expf(s0 - m);
  float p1 = (lane == 0) ? __expf(s1 - m) : 0.f;
  float sum = p0 + p1;
#pragma unroll
  for (int off = 16; off > 0; off >>= 1) sum += __shfl_xor(sum, off, 32);
  const float inv = 1.0f / sum;
  attn_s[h][lane] = p0 * inv;
  if (lane == 0) attn_s[h][32] = p1 * inv;
  __syncthreads();

  float o0 = 0.f, o1 = 0.f;
  for (int w = 0; w < W_; ++w) {
    const float aw = attn_s[h][w];
    o0 += aw * bf2f(Vp[w * 520 + h * DK_ + lane]);
    o1 += aw * bf2f(Vp[w * 520 + h * DK_ + lane + 32]);
  }
  xout[(long)bt * F_ + h * DK_ + lane]      = o0;
  xout[(long)bt * F_ + h * DK_ + lane + 32] = o1;
}

extern "C" void kernel_launch(void* const* d_in, const int* in_sizes, int n_in,
                              void* d_out, int out_size, void* d_ws, size_t ws_size,
                              hipStream_t stream) {
  const float* query = (const float*)d_in[0];
  const float* key   = (const float*)d_in[1];
  const float* value = (const float*)d_in[2];
  const int*   mask  = (const int*)d_in[3];
  const float* Wq = (const float*)d_in[4];
  const float* bq = (const float*)d_in[5];
  const float* Wk = (const float*)d_in[6];
  const float* bk = (const float*)d_in[7];
  const float* Wv = (const float*)d_in[8];
  const float* bv = (const float*)d_in[9];
  const float* Wo = (const float*)d_in[10];
  const float* bo = (const float*)d_in[11];
  float* out = (float*)d_out;

  char* ws = (char*)d_ws;
  const size_t WB = (size_t)F_ * F_ * sizeof(unsigned short);          // 512 KB / weight
  unsigned short* wqT = (unsigned short*)(ws);
  unsigned short* wkT = (unsigned short*)(ws + WB);
  unsigned short* wvT = (unsigned short*)(ws + 2 * WB);
  unsigned short* woT = (unsigned short*)(ws + 3 * WB);
  float* qp = (float*)(ws + 4 * WB);                                   // 4096x512 fp32
  float* xo = (float*)(ws + 4 * WB + (size_t)B_ * T_ * F_ * sizeof(float));

  cvt_transpose<<<256, 256, 0, stream>>>(Wq, wqT);
  cvt_transpose<<<256, 256, 0, stream>>>(Wk, wkT);
  cvt_transpose<<<256, 256, 0, stream>>>(Wv, wvT);
  cvt_transpose<<<256, 256, 0, stream>>>(Wo, woT);

  dim3 gq(B_ * T_ / 64, F_ / 64);
  gemm_bias<<<gq, 256, 0, stream>>>(query, wqT, bq, qp);               // Q projection
  fused_attn<<<B_ * T_, 256, 0, stream>>>(key, value, mask, wkT, bk, wvT, bv, qp, xo);
  gemm_bias<<<gq, 256, 0, stream>>>(xo, woT, bo, out);                 // O projection
}
